// End2End_IP_56779467653589
// MI455X (gfx1250) — compile-verified, matrix operand order checked by reading, not run on hardware
//
#include <hip/hip_runtime.h>
#include <hip/hip_bf16.h>

#define NUM_ELEMENTS 20000
#define NUM_CONCEPTS 512
#define IN_CHANNELS  768
#define NUM_CLASSES  1000
#define BATCH        16384
#define FEAT_IDX     256
#define LN_EPS       1e-5f

typedef __attribute__((ext_vector_type(16))) __bf16 v16bf;
typedef __attribute__((ext_vector_type(8)))  float  v8f;

__device__ __forceinline__ unsigned short f2bf(float f) {
    unsigned int u = __float_as_uint(f);
    unsigned int r = u + 0x7FFFu + ((u >> 16) & 1u);   // round-to-nearest-even
    return (unsigned short)(r >> 16);
}

// Deterministic Gumbel noise: splitmix64 counter hash -> uniform (0,1) -> -log(-log(u))
__device__ __forceinline__ float gumbel_noise(int e, int c) {
    unsigned long long z = (unsigned long long)e * NUM_CONCEPTS + (unsigned long long)c
                         + 42ull * 0xD6E8FEB86659FD93ull;
    z += 0x9E3779B97F4A7C15ull;
    z = (z ^ (z >> 30)) * 0xBF58476D1CE4E5B9ull;
    z = (z ^ (z >> 27)) * 0x94D049BB133111EBull;
    z = z ^ (z >> 31);
    float u = ((float)(z >> 40) + 0.5f) * (1.0f / 16777216.0f);  // (0,1)
    return -__logf(-__logf(u));
}

// ---------------- zero workspace ----------------
__global__ void k_zero(float* p, int n) {
    int i = blockIdx.x * blockDim.x + threadIdx.x;
    if (i < n) p[i] = 0.0f;
}

// ---------------- per-element: layernorm(q)*10 + gumbel -> argmax -> scatter-add sc ----------------
__global__ __launch_bounds__(256) void k_assign(const float* __restrict__ q,
                                                const float* __restrict__ qw,
                                                const float* __restrict__ qb,
                                                const float* __restrict__ sc,
                                                float* __restrict__ proto) {
    __shared__ float red[256];
    __shared__ int   redi[256];
    const int e = blockIdx.x;
    const int t = threadIdx.x;
    const float* qr = q + (size_t)e * NUM_CONCEPTS;
    float v0 = qr[t], v1 = qr[t + 256];

    red[t] = v0 + v1; __syncthreads();
    for (int s = 128; s > 0; s >>= 1) { if (t < s) red[t] += red[t + s]; __syncthreads(); }
    float mu = red[0] * (1.0f / NUM_CONCEPTS); __syncthreads();

    float d0 = v0 - mu, d1 = v1 - mu;
    red[t] = d0 * d0 + d1 * d1; __syncthreads();
    for (int s = 128; s > 0; s >>= 1) { if (t < s) red[t] += red[t + s]; __syncthreads(); }
    float rs = rsqrtf(red[0] * (1.0f / NUM_CONCEPTS) + LN_EPS); __syncthreads();

    float l0 = (d0 * rs * qw[t]       + qb[t])       * 10.0f + gumbel_noise(e, t);
    float l1 = (d1 * rs * qw[t + 256] + qb[t + 256]) * 10.0f + gumbel_noise(e, t + 256);
    float bv = (l1 > l0) ? l1 : l0;
    int   bi = (l1 > l0) ? (t + 256) : t;
    red[t] = bv; redi[t] = bi; __syncthreads();
    for (int s = 128; s > 0; s >>= 1) {
        if (t < s) {
            float o = red[t + s]; int oi = redi[t + s];
            if (o > red[t] || (o == red[t] && oi < redi[t])) { red[t] = o; redi[t] = oi; }
        }
        __syncthreads();
    }
    const int idx = redi[0];

    const float* scr = sc + (size_t)e * IN_CHANNELS;
    float* pr = proto + (size_t)idx * IN_CHANNELS;
    for (int c = t; c < IN_CHANNELS; c += 256) atomicAdd(&pr[c], scr[c]);
}

// ---------------- L2-normalize prototype rows, emit bf16 ----------------
__global__ __launch_bounds__(256) void k_pnorm(const float* __restrict__ proto,
                                               unsigned short* __restrict__ p16) {
    __shared__ float red[256];
    const int r = blockIdx.x, t = threadIdx.x;
    const float* pr = proto + (size_t)r * IN_CHANNELS;
    float a = pr[t], b = pr[t + 256], c = pr[t + 512];
    red[t] = a * a + b * b + c * c; __syncthreads();
    for (int s = 128; s > 0; s >>= 1) { if (t < s) red[t] += red[t + s]; __syncthreads(); }
    float inv = 1.0f / fmaxf(sqrtf(red[0]), 1e-12f);
    unsigned short* o = p16 + (size_t)r * IN_CHANNELS;
    o[t]       = f2bf(a * inv);
    o[t + 256] = f2bf(b * inv);
    o[t + 512] = f2bf(c * inv);
}

// ---------------- f32 -> bf16 convert (fc_w only) ----------------
__global__ void k_cvt(const float* __restrict__ in, unsigned short* __restrict__ out, int n) {
    int i = blockIdx.x * blockDim.x + threadIdx.x;
    int stride = gridDim.x * blockDim.x;
    for (; i < n; i += stride) out[i] = f2bf(in[i]);
}

// =====================================================================================
// Fused: GEMM1 (x @ P^T) -> layernorm -> median-abs threshold mask -> GEMM2 (@ fc_w^T)
// One block handles 16 full output rows; no f32/bf16 feature round-trips to HBM.
// =====================================================================================
#define XS 776   // bf16 LDS stride for x tile (768 + 8 pad -> de-banked A-frag reads)
#define YS 516   // f32  LDS stride for y tile (512 + 4 pad)
#define CS 520   // bf16 LDS stride for masked-feature tile (512 + 8 pad)

__global__ __launch_bounds__(256) void k_fused(const float* __restrict__ x,
                                               const unsigned short* __restrict__ P16,
                                               const float* __restrict__ fw,
                                               const float* __restrict__ fb,
                                               const unsigned short* __restrict__ W16,
                                               const float* __restrict__ bias,
                                               float* __restrict__ out) {
    __shared__ __align__(32) unsigned short xl[16 * XS];   // 24,832 B (reused for cf16 tile)
    __shared__ __align__(32) float          yl[16 * YS];   // 33,024 B
    __shared__ __align__(32) float          sl[16 * 512];  // 32,768 B

    const int t    = threadIdx.x;
    const int wave = t >> 5;
    const int lane = t & 31;
    const int half = lane >> 4, r = lane & 15;
    const int m0   = blockIdx.x * 16;

    // ---- stage 1: load x tile (16 x 768 f32), convert to bf16 in LDS ----
    for (int i = t; i < 16 * IN_CHANNELS; i += 256) {
        int rr = i / IN_CHANNELS, cc = i - rr * IN_CHANNELS;
        xl[rr * XS + cc] = f2bf(x[(size_t)(m0 + rr) * IN_CHANNELS + cc]);
    }
    __syncthreads();

    // ---- stage 2: WMMA GEMM1: wave computes 16x64, block covers full N=512 ----
    {
        const int n0 = wave * 64;
        const unsigned short* al = &xl[r * XS + half * 16];
        v8f acc0 = {}, acc1 = {}, acc2 = {}, acc3 = {};
        for (int k0 = 0; k0 < IN_CHANNELS; k0 += 32) {
            v16bf a = *(const v16bf*)(al + k0);
            const unsigned short* bbase = P16 + (size_t)(n0 + r) * IN_CHANNELS + k0 + half * 16;
            v16bf b0 = *(const v16bf*)(bbase);
            v16bf b1 = *(const v16bf*)(bbase + (size_t)16 * IN_CHANNELS);
            v16bf b2 = *(const v16bf*)(bbase + (size_t)32 * IN_CHANNELS);
            v16bf b3 = *(const v16bf*)(bbase + (size_t)48 * IN_CHANNELS);
            acc0 = __builtin_amdgcn_wmma_f32_16x16x32_bf16(false, a, false, b0, (short)0, acc0, false, false);
            acc1 = __builtin_amdgcn_wmma_f32_16x16x32_bf16(false, a, false, b1, (short)0, acc1, false, false);
            acc2 = __builtin_amdgcn_wmma_f32_16x16x32_bf16(false, a, false, b2, (short)0, acc2, false, false);
            acc3 = __builtin_amdgcn_wmma_f32_16x16x32_bf16(false, a, false, b3, (short)0, acc3, false, false);
        }
        v8f accs[4] = {acc0, acc1, acc2, acc3};
        #pragma unroll
        for (int j = 0; j < 4; j++)
            #pragma unroll
            for (int v = 0; v < 8; v++)
                yl[(half * 8 + v) * YS + n0 + j * 16 + r] = accs[j][v];
    }
    __syncthreads();

    // ---- stage 3: per-row layernorm (16 lanes per row, shfl reductions) ----
    {
        const int rr = t >> 4, lt = t & 15;         // rows 2w,2w+1 live in wave w -> shfl ok
        float vals[32];
        float s = 0.0f;
        #pragma unroll
        for (int ii = 0; ii < 32; ii++) { vals[ii] = yl[rr * YS + lt + 16 * ii]; s += vals[ii]; }
        #pragma unroll
        for (int m = 1; m < 16; m <<= 1) s += __shfl_xor(s, m, 32);
        float mu = s * (1.0f / NUM_CONCEPTS);
        float vs = 0.0f;
        #pragma unroll
        for (int ii = 0; ii < 32; ii++) { float d = vals[ii] - mu; vs += d * d; }
        #pragma unroll
        for (int m = 1; m < 16; m <<= 1) vs += __shfl_xor(vs, m, 32);
        float rsf = rsqrtf(vs * (1.0f / NUM_CONCEPTS) + LN_EPS);
        #pragma unroll
        for (int ii = 0; ii < 32; ii++) {
            int c = lt + 16 * ii;
            float y = (vals[ii] - mu) * rsf * fw[c] + fb[c];
            yl[rr * YS + c]   = y;
            sl[(rr << 9) + c] = fabsf(y);
        }
    }
    __syncthreads();

    // ---- stage 4: bitonic sort |y| for all 16 rows (512 each), block-cooperative ----
    for (int k = 2; k <= 512; k <<= 1) {
        for (int j = k >> 1; j > 0; j >>= 1) {
            for (int e = t; e < 16 * 512; e += 256) {
                int row = e >> 9, idx = e & 511;
                int ixj = idx ^ j;
                if (ixj > idx) {
                    float* base = sl + (row << 9);
                    float a = base[idx], b = base[ixj];
                    bool up = ((idx & k) == 0);
                    if ((a > b) == up) { base[idx] = b; base[ixj] = a; }
                }
            }
            __syncthreads();
        }
    }

    // ---- stage 5: threshold-mask, write masked features as bf16 tile (reuse xl) ----
    for (int i = t; i < 16 * NUM_CONCEPTS; i += 256) {
        int row = i >> 9, c = i & 511;
        float y = yl[row * YS + c];
        float thres = sl[(row << 9) + FEAT_IDX];
        xl[row * CS + c] = f2bf((fabsf(y) > thres) ? y : 0.0f);
    }
    __syncthreads();

    // ---- stage 6: WMMA GEMM2: features(16x512) @ fc_w^T(512x1000) + bias ----
    {
        const unsigned short* al = &xl[r * CS + half * 16];
        for (int s6 = 0; s6 < 8; s6++) {
            const int n0 = (wave + 8 * s6) * 16;          // 64 tiles cover 1024 cols
            int nb = n0 + r; if (nb > NUM_CLASSES - 1) nb = NUM_CLASSES - 1;
            const unsigned short* bptr = W16 + (size_t)nb * NUM_CONCEPTS + half * 16;
            v8f acc = {};
            for (int k0 = 0; k0 < NUM_CONCEPTS; k0 += 32) {
                v16bf a = *(const v16bf*)(al + k0);
                v16bf b = *(const v16bf*)(bptr + k0);
                acc = __builtin_amdgcn_wmma_f32_16x16x32_bf16(false, a, false, b, (short)0, acc, false, false);
            }
            const int n = n0 + r;
            if (n < NUM_CLASSES) {
                float bv = bias[n];
                float* cp = out + (size_t)(m0 + half * 8) * NUM_CLASSES + n;
                #pragma unroll
                for (int v = 0; v < 8; v++) cp[(size_t)v * NUM_CLASSES] = acc[v] + bv;
            }
        }
    }
}

extern "C" void kernel_launch(void* const* d_in, const int* in_sizes, int n_in,
                              void* d_out, int out_size, void* d_ws, size_t ws_size,
                              hipStream_t stream) {
    const float* x   = (const float*)d_in[0];
    const float* sc  = (const float*)d_in[1];
    const float* q   = (const float*)d_in[2];
    const float* qw  = (const float*)d_in[3];
    const float* qb  = (const float*)d_in[4];
    const float* fw  = (const float*)d_in[5];
    const float* fb  = (const float*)d_in[6];
    const float* fcw = (const float*)d_in[7];
    const float* fcb = (const float*)d_in[8];
    float* out = (float*)d_out;

    char* ws = (char*)d_ws;
    float*          proto = (float*)(ws + 0);                 // 512*768*4  = 1,572,864
    unsigned short* p16   = (unsigned short*)(ws + 1572864);  // 512*768*2  =   786,432
    unsigned short* fcw16 = (unsigned short*)(ws + 2359296);  // 1000*512*2 = 1,024,000
    // total workspace: ~3.4 MB

    const int nproto = NUM_CONCEPTS * IN_CHANNELS;            // 393,216
    k_zero<<<(nproto + 255) / 256, 256, 0, stream>>>(proto, nproto);

    k_assign<<<NUM_ELEMENTS, 256, 0, stream>>>(q, qw, qb, sc, proto);

    k_pnorm<<<NUM_CONCEPTS, 256, 0, stream>>>(proto, p16);

    const int nf = NUM_CLASSES * NUM_CONCEPTS;                // 512,000
    k_cvt<<<2000, 256, 0, stream>>>(fcw, fcw16, nf);

    k_fused<<<BATCH / 16, 256, 0, stream>>>(x, p16, fw, fb, fcw16, fcb, out);
}
